// RPN_40080634806596
// MI455X (gfx1250) — matrix operand order repaired and checked by reading, not compiled, not run on hardware
//
#include <hip/hip_runtime.h>
#include <stdint.h>

#define H_  100
#define W_  150
#define CIN 1024
#define CMID 512
#define KTOT (9*CIN)        // 9216
#define NPOS (H_*W_)        // 15000
#define NANCH (NPOS*9)      // 135000
#define PRE_NMS 6000
#define POST_NMS 300
#define NMS_T 0.7f
#define MIN_SIZE 16.0f
#define SORT_N 262144       // next pow2 >= 135000
#define NMS_WORDS 188       // ceil(6000/32)

typedef __attribute__((ext_vector_type(16))) __bf16 bf16x16;
typedef __attribute__((ext_vector_type(8)))  float  v8f;

union Frag { uint4 q[2]; bf16x16 v; };

static __device__ __forceinline__ unsigned short f2bf(float f) {
    unsigned u = __float_as_uint(f);
    u += 0x7FFFu + ((u >> 16) & 1u);   // round-to-nearest-even
    return (unsigned short)(u >> 16);
}

// ---------------- 1) pad + convert feature map to bf16 (zero halo) ----------
__global__ void k_pad_bf16(const float* __restrict__ feat, unsigned short* __restrict__ Fp) {
    int i = blockIdx.x * 256 + threadIdx.x;
    const int total = 102 * 152 * CIN;
    if (i >= total) return;
    int c  = i % CIN;
    int sp = i / CIN;
    int xx = sp % 152, yy = sp / 152;
    unsigned short v = 0;
    if (yy >= 1 && yy <= H_ && xx >= 1 && xx <= W_)
        v = f2bf(feat[((size_t)(yy - 1) * W_ + (xx - 1)) * CIN + c]);
    Fp[i] = v;
}

// ------- 2) transpose+convert weights: Wr[k][co] f32 -> Wt[co][k] bf16 ------
__global__ __launch_bounds__(256) void k_w_bf16t(const float* __restrict__ Wr,
                                                 unsigned short* __restrict__ Wt) {
    __shared__ float tile[32][33];
    int k0 = blockIdx.x * 32;         // K tiles: 9216/32 = 288
    int c0 = blockIdx.y * 32;         // C tiles: 512/32 = 16
    int tx = threadIdx.x & 31;
    int ty = threadIdx.x >> 5;        // 0..7
    #pragma unroll
    for (int i = 0; i < 4; ++i) {
        int kk = ty * 4 + i;
        tile[kk][tx] = Wr[(size_t)(k0 + kk) * CMID + c0 + tx];   // coalesced read
    }
    __syncthreads();
    #pragma unroll
    for (int i = 0; i < 4; ++i) {
        int cc = ty * 4 + i;
        Wt[(size_t)(c0 + cc) * KTOT + k0 + tx] = f2bf(tile[tx][cc]);  // coalesced write
    }
}

// ---------------- 3) 3x3 conv as implicit GEMM via WMMA bf16 ----------------
// M=15000, N=512, K=9216. Block=128 thr (4 waves), tile 64x128, Kstep=32.
// Tiles move global->LDS via GLOBAL_LOAD_ASYNC_TO_LDS_B128 (no VGPR staging),
// double-buffered, one s_wait_asynccnt + barrier per K-step, 8 WMMAs/wave/step.
__global__ __launch_bounds__(128, 1) void k_conv3x3(
        const unsigned short* __restrict__ Fp, const unsigned short* __restrict__ Wt,
        const float* __restrict__ brpn, float* __restrict__ rpn) {
    __shared__ __align__(16) unsigned short As[2][64 * 32];    // [m][k]
    __shared__ __align__(16) unsigned short Bs[2][128 * 32];   // [n][k]
    const int tid  = threadIdx.x;
    const int lane = tid & 31;
    const int wv   = tid >> 5;
    const int wM   = wv >> 1, wN = wv & 1;
    const int tM   = blockIdx.x, tN = blockIdx.y;

    // per-thread A tile slots (2 x b128): hoisted row bases.
    // OOB rows (m >= NPOS) are clamped to a valid row; their outputs are
    // discarded by the epilogue guard, so garbage contributions are harmless.
    int aoff[2], aldst[2];
    #pragma unroll
    for (int it = 0; it < 2; ++it) {
        int s = tid + it * 128;
        int r = s >> 2, qi = s & 3;
        int m = tM * 64 + r;
        int mc = (m < NPOS) ? m : (NPOS - 1);
        int y = mc / W_, x = mc - y * W_;
        aldst[it] = r * 32 + qi * 8;
        aoff[it]  = (y * 152 + x) * CIN + qi * 8;   // tap(0,0) corner base
    }
    // per-thread B tile slots (4 x b128)
    size_t boff[4]; int bldst[4];
    #pragma unroll
    for (int it = 0; it < 4; ++it) {
        int s = tid + it * 128;
        int n = s >> 2, qi = s & 3;
        bldst[it] = n * 32 + qi * 8;
        boff[it] = (size_t)(tN * 128 + n) * KTOT + qi * 8;
    }

    v8f acc[2][4] = {};
    const int NSTEP = KTOT / 32;   // 288

    // async DMA of one K-step tile pair into LDS buffer `buf` (ASYNCcnt)
    auto issue_async = [&](int step, int buf) {
        int kyx = step >> 5;                    // tap index 0..8
        int ky  = (kyx * 11) >> 5;              // /3 for 0..8
        int kx  = kyx - ky * 3;
        int kofs = (ky * 152 + kx) * CIN + (step & 31) * 32;  // wave-uniform
        int ks   = step * 32;
        #pragma unroll
        for (int it = 0; it < 2; ++it) {
            const void* g = (const void*)(Fp + (size_t)(aoff[it] + kofs));
            unsigned l = (unsigned)(uintptr_t)&As[buf][aldst[it]];
            asm volatile("global_load_async_to_lds_b128 %0, %1, off"
                         :: "v"(l), "v"(g) : "memory");
        }
        #pragma unroll
        for (int it = 0; it < 4; ++it) {
            const void* g = (const void*)(Wt + boff[it] + ks);
            unsigned l = (unsigned)(uintptr_t)&Bs[buf][bldst[it]];
            asm volatile("global_load_async_to_lds_b128 %0, %1, off"
                         :: "v"(l), "v"(g) : "memory");
        }
    };

    issue_async(0, 0);
    asm volatile("s_wait_asynccnt 0x0" ::: "memory");
    __syncthreads();

    // fragment addressing per ISA 16-bit layouts
    const int k0 = (lane < 16) ? 0 : 8;     // A: K runs {k0..k0+7, k0+16..k0+23}
    const int kb = (lane < 16) ? 0 : 8;     // B: K runs {kb..kb+7, kb+16..kb+23}
    const int am = wM * 32 + (lane & 15);
    const int bn = wN * 64 + (lane & 15);

    for (int step = 0; step < NSTEP; ++step) {
        int cur = step & 1;
        // buffer cur^1's last readers finished before the previous barrier
        if (step + 1 < NSTEP) issue_async(step + 1, cur ^ 1);
        Frag fa[2], fb[4];
        #pragma unroll
        for (int sm = 0; sm < 2; ++sm) {
            int r = am + sm * 16;
            fa[sm].q[0] = *(const uint4*)(&As[cur][r * 32 + k0]);
            fa[sm].q[1] = *(const uint4*)(&As[cur][r * 32 + k0 + 16]);
        }
        #pragma unroll
        for (int sn = 0; sn < 4; ++sn) {
            int n = bn + sn * 16;
            fb[sn].q[0] = *(const uint4*)(&Bs[cur][n * 32 + kb]);
            fb[sn].q[1] = *(const uint4*)(&Bs[cur][n * 32 + kb + 16]);
        }
        #pragma unroll
        for (int sm = 0; sm < 2; ++sm)
            #pragma unroll
            for (int sn = 0; sn < 4; ++sn)
                acc[sm][sn] = __builtin_amdgcn_wmma_f32_16x16x32_bf16(
                    false, fa[sm].v, false, fb[sn].v, (short)0, acc[sm][sn], false, false);
        if (step + 1 < NSTEP)
            asm volatile("s_wait_asynccnt 0x0" ::: "memory");
        __syncthreads();
    }

    // epilogue: bias + ReLU; C/D layout: VGPR j -> row j + 8*(lane>=16), col lane&15
    int half = (lane >> 4) & 1;
    #pragma unroll
    for (int sm = 0; sm < 2; ++sm) {
        #pragma unroll
        for (int sn = 0; sn < 4; ++sn) {
            int co = tN * 128 + wN * 64 + sn * 16 + (lane & 15);
            float bias = brpn[co];
            #pragma unroll
            for (int j = 0; j < 8; ++j) {
                int m = tM * 64 + wM * 32 + sm * 16 + j + half * 8;
                if (m < NPOS) {
                    float v = acc[sm][sn][j] + bias;
                    rpn[(size_t)m * CMID + co] = v > 0.f ? v : 0.f;
                }
            }
        }
    }
}

// ---------------- 4) 1x1 heads + softmax + anchor decode --------------------
__device__ __constant__ float ANCH[9][4] = {
    {-84.f,-40.f,99.f,55.f},   {-176.f,-88.f,191.f,103.f}, {-360.f,-184.f,375.f,199.f},
    {-56.f,-56.f,71.f,71.f},   {-120.f,-120.f,135.f,135.f},{-248.f,-248.f,263.f,263.f},
    {-36.f,-80.f,51.f,95.f},   {-80.f,-168.f,95.f,183.f},  {-168.f,-352.f,183.f,367.f}};

__global__ __launch_bounds__(64) void k_heads(
        const float* __restrict__ rpn,
        const float* __restrict__ Wsc, const float* __restrict__ bsc,
        const float* __restrict__ Wbb, const float* __restrict__ bbb,
        const int* __restrict__ img,
        float* __restrict__ scores, float* __restrict__ boxes) {
    __shared__ float row[CMID];
    __shared__ float outs[54];
    int p = blockIdx.x, t = threadIdx.x;
    #pragma unroll
    for (int j = 0; j < 8; ++j) row[t * 8 + j] = rpn[(size_t)p * CMID + t * 8 + j];
    __syncthreads();
    if (t < 54) {
        float acc;
        if (t < 18) {
            acc = bsc[t];
            for (int ci = 0; ci < CMID; ++ci) acc += row[ci] * Wsc[ci * 18 + t];
        } else {
            int co = t - 18;
            acc = bbb[co];
            for (int ci = 0; ci < CMID; ++ci) acc += row[ci] * Wbb[ci * 36 + co];
        }
        outs[t] = acc;
    }
    __syncthreads();
    if (t < 9) {
        int a = t;
        float c0 = outs[2 * a], c1 = outs[2 * a + 1];
        float fg = 1.f / (1.f + expf(c0 - c1));          // 2-way softmax fg prob
        float d0 = outs[18 + 4 * a], d1 = outs[18 + 4 * a + 1];
        float d2 = outs[18 + 4 * a + 2], d3 = outs[18 + 4 * a + 3];
        int y = p / W_, x = p - y * W_;
        float sx = x * 16.f, sy = y * 16.f;
        float ax1 = ANCH[a][0] + sx, ay1 = ANCH[a][1] + sy;
        float ax2 = ANCH[a][2] + sx, ay2 = ANCH[a][3] + sy;
        float aw = ax2 - ax1 + 1.f, ah = ay2 - ay1 + 1.f;
        float acx = ax1 + 0.5f * aw, acy = ay1 + 0.5f * ah;
        float pcx = d0 * aw + acx, pcy = d1 * ah + acy;
        float pw = expf(d2) * aw, ph = expf(d3) * ah;
        float imh = (float)img[0], imw = (float)img[1];
        float x1 = fminf(fmaxf(pcx - 0.5f * pw, 0.f), imw - 1.f);
        float y1 = fminf(fmaxf(pcy - 0.5f * ph, 0.f), imh - 1.f);
        float x2 = fminf(fmaxf(pcx + 0.5f * pw, 0.f), imw - 1.f);
        float y2 = fminf(fmaxf(pcy + 0.5f * ph, 0.f), imh - 1.f);
        float bw = x2 - x1 + 1.f, bh = y2 - y1 + 1.f;
        if (!(bw >= MIN_SIZE && bh >= MIN_SIZE)) fg = -1e9f;
        int idx = p * 9 + a;
        scores[idx] = fg;
        float* bo = boxes + (size_t)idx * 4;
        bo[0] = x1; bo[1] = y1; bo[2] = x2; bo[3] = y2;
    }
}

// ---------------- 5) sortable keys (desc score, asc index ties) -------------
__global__ void k_keys(const float* __restrict__ scores, unsigned long long* __restrict__ keys) {
    int i = blockIdx.x * 256 + threadIdx.x;
    if (i >= SORT_N) return;
    unsigned long long k = 0ull;    // padding sorts to the end
    if (i < NANCH) {
        unsigned u = __float_as_uint(scores[i]);
        u = (u & 0x80000000u) ? ~u : (u | 0x80000000u);   // order-preserving map
        k = ((unsigned long long)u << 32) | (unsigned)(~(unsigned)i);
    }
    keys[i] = k;
}

__global__ void k_bitonic(unsigned long long* __restrict__ keys, unsigned k, unsigned j) {
    unsigned i = blockIdx.x * 256 + threadIdx.x;
    unsigned ixj = i ^ j;
    if (ixj > i) {
        unsigned long long a = keys[i], b = keys[ixj];
        bool desc = ((i & k) == 0u);
        if (desc ? (a < b) : (a > b)) { keys[i] = b; keys[ixj] = a; }
    }
}

__global__ void k_gather(const unsigned long long* __restrict__ keys,
                         const float* __restrict__ scores, const float* __restrict__ boxes,
                         float* __restrict__ tops, float* __restrict__ topb) {
    int i = blockIdx.x * 256 + threadIdx.x;
    if (i >= PRE_NMS) return;
    unsigned idx = ~(unsigned)(keys[i] & 0xFFFFFFFFull);
    float* o = topb + (size_t)i * 4;
    if (idx < NANCH) {
        tops[i] = scores[idx];
        const float* b = boxes + (size_t)idx * 4;
        o[0] = b[0]; o[1] = b[1]; o[2] = b[2]; o[3] = b[3];
    } else {
        tops[i] = -1e9f; o[0] = o[1] = o[2] = o[3] = 0.f;
    }
}

// ---------------- 6) NMS: parallel suppression bitmatrix --------------------
__global__ void k_nms_mask(const float* __restrict__ topb, unsigned* __restrict__ mask) {
    int id = blockIdx.x * 256 + threadIdx.x;
    if (id >= PRE_NMS * NMS_WORDS) return;
    int i = id / NMS_WORDS, c = id - i * NMS_WORDS;
    const float* bi = topb + (size_t)i * 4;
    float x1 = bi[0], y1 = bi[1], x2 = bi[2], y2 = bi[3];
    float ai = (x2 - x1 + 1.f) * (y2 - y1 + 1.f);
    unsigned m = 0u;
    int j0 = c * 32;
    #pragma unroll 4
    for (int b = 0; b < 32; ++b) {
        int j = j0 + b;
        if (j >= PRE_NMS || j <= i) continue;
        const float* bj = topb + (size_t)j * 4;
        float iw = fminf(x2, bj[2]) - fmaxf(x1, bj[0]) + 1.f;
        float ih = fminf(y2, bj[3]) - fmaxf(y1, bj[1]) + 1.f;
        iw = fmaxf(iw, 0.f); ih = fmaxf(ih, 0.f);
        float inter = iw * ih;
        float aj = (bj[2] - bj[0] + 1.f) * (bj[3] - bj[1] + 1.f);
        float iou = inter / (ai + aj - inter);
        if (iou > NMS_T) m |= (1u << b);
    }
    mask[(size_t)i * NMS_WORDS + c] = m;
}

// ---------------- 7) greedy merge (single block) ----------------------------
__global__ __launch_bounds__(256) void k_nms_merge(const unsigned* __restrict__ mask,
                                                   unsigned* __restrict__ keep) {
    __shared__ unsigned rem[NMS_WORDS];
    int t = threadIdx.x;
    if (t < NMS_WORDS) rem[t] = 0u;
    __syncthreads();
    for (int i = 0; i < PRE_NMS; ++i) {
        unsigned wrd = rem[i >> 5];
        __syncthreads();
        bool alive = ((wrd >> (i & 31)) & 1u) == 0u;
        if (alive && t < NMS_WORDS) rem[t] |= mask[(size_t)i * NMS_WORDS + t];
        if (t == 0) keep[i] = alive ? 1u : 0u;
        __syncthreads();
    }
}

// ---------------- 8) compact first 300 kept ---------------------------------
__global__ __launch_bounds__(256) void k_out(const unsigned* __restrict__ keep,
                                             const float* __restrict__ tops,
                                             const float* __restrict__ topb,
                                             float* __restrict__ out) {
    int t = threadIdx.x;
    for (int i = t; i < POST_NMS * 4; i += 256) out[i] = 0.f;
    __syncthreads();
    if (t == 0) {
        int cnt = 0;
        for (int i = 0; i < PRE_NMS && cnt < POST_NMS; ++i) {
            if (keep[i] && tops[i] > -1e8f) {
                out[cnt * 4 + 0] = topb[i * 4 + 0];
                out[cnt * 4 + 1] = topb[i * 4 + 1];
                out[cnt * 4 + 2] = topb[i * 4 + 2];
                out[cnt * 4 + 3] = topb[i * 4 + 3];
                ++cnt;
            }
        }
    }
}

extern "C" void kernel_launch(void* const* d_in, const int* in_sizes, int n_in,
                              void* d_out, int out_size, void* d_ws, size_t ws_size,
                              hipStream_t stream) {
    (void)in_sizes; (void)n_in; (void)out_size; (void)ws_size;
    const float* feat = (const float*)d_in[0];
    const float* Wr   = (const float*)d_in[1];
    const float* br   = (const float*)d_in[2];
    const float* Wsc  = (const float*)d_in[3];
    const float* bsc  = (const float*)d_in[4];
    const float* Wbb  = (const float*)d_in[5];
    const float* bbb  = (const float*)d_in[6];
    const int*   img  = (const int*)d_in[7];
    float* out = (float*)d_out;

    char* base = (char*)d_ws;
    size_t off = 0;
    auto alloc = [&](size_t bytes) -> void* {
        void* p = base + off;
        off = (off + bytes + 255) & ~(size_t)255;
        return p;
    };
    unsigned short*     Fp     = (unsigned short*)alloc((size_t)102 * 152 * CIN * 2);
    unsigned short*     Wt     = (unsigned short*)alloc((size_t)CMID * KTOT * 2);
    float*              rpn    = (float*)alloc((size_t)NPOS * CMID * 4);
    float*              scores = (float*)alloc((size_t)NANCH * 4);
    float*              boxes  = (float*)alloc((size_t)NANCH * 4 * 4);
    unsigned long long* keys   = (unsigned long long*)alloc((size_t)SORT_N * 8);
    float*              tops   = (float*)alloc((size_t)PRE_NMS * 4);
    float*              topb   = (float*)alloc((size_t)PRE_NMS * 4 * 4);
    unsigned*           mask   = (unsigned*)alloc((size_t)PRE_NMS * NMS_WORDS * 4);
    unsigned*           keep   = (unsigned*)alloc((size_t)PRE_NMS * 4);

    k_pad_bf16<<<(102 * 152 * CIN + 255) / 256, 256, 0, stream>>>(feat, Fp);
    dim3 gw(KTOT / 32, CMID / 32);
    k_w_bf16t<<<gw, 256, 0, stream>>>(Wr, Wt);
    dim3 g3((NPOS + 63) / 64, CMID / 128);
    k_conv3x3<<<g3, 128, 0, stream>>>(Fp, Wt, br, rpn);
    k_heads<<<NPOS, 64, 0, stream>>>(rpn, Wsc, bsc, Wbb, bbb, img, scores, boxes);
    k_keys<<<SORT_N / 256, 256, 0, stream>>>(scores, keys);
    for (unsigned k = 2; k <= SORT_N; k <<= 1)
        for (unsigned j = k >> 1; j > 0; j >>= 1)
            k_bitonic<<<SORT_N / 256, 256, 0, stream>>>(keys, k, j);
    k_gather<<<(PRE_NMS + 255) / 256, 256, 0, stream>>>(keys, scores, boxes, tops, topb);
    k_nms_mask<<<(PRE_NMS * NMS_WORDS + 255) / 256, 256, 0, stream>>>(topb, mask);
    k_nms_merge<<<1, 256, 0, stream>>>(mask, keep);
    k_out<<<1, 256, 0, stream>>>(keep, tops, topb, out);
}
